// GPT_7962869367573
// MI455X (gfx1250) — compile-verified, matrix operand order checked by reading, not run on hardware
//
#include <hip/hip_runtime.h>
#include <math.h>

// ---------------------------------------------------------------------------
// GPT forward for MI455X (gfx1250, wave32, WMMA + TDM).
// - All matmuls on v_wmma_f32_16x16x32_bf16 (f32 accumulate).
// - Weights converted f32->bf16 AND pre-transposed once per launch, so both
//   GEMM operands stage as row-major 128x64 tiles -> staged by the Tensor
//   Data Mover (tensor_load_to_lds, padded LDS layout), fragments load as
//   2x ds_load_b128 in the documented 16-bit WMMA K-striping.
// - Flash attention (no TxT buffer), online softmax, WMMA for QK^T and PV.
// Requires ws_size >= ~192MB.
// ---------------------------------------------------------------------------

typedef __bf16 v16bf __attribute__((ext_vector_type(16)));
typedef float  v8f   __attribute__((ext_vector_type(8)));
typedef unsigned int u32x4 __attribute__((ext_vector_type(4)));
typedef unsigned int u32x2 __attribute__((ext_vector_type(2)));
typedef int    i32x4 __attribute__((ext_vector_type(4)));
typedef int    i32x8 __attribute__((ext_vector_type(8)));
typedef float  f32x4 __attribute__((ext_vector_type(4)));

union Frag16 { v16bf v; u32x4 q2[2]; unsigned short u[16]; };
union U4U8   { u32x4 q; unsigned short s[8]; };

#define Bsz 2
#define Tdim 2048
#define Edim 1024
#define Hn   16
#define Ddim 64
#define Fdim 4096
#define Vdim 32000
#define Ldim 4
#define Mrows (Bsz*Tdim)   // 4096

// --- toolchain feature detection -------------------------------------------
#if defined(__has_include)
# if __has_include(<hip/amd_detail/amd_gfx1250_TDM.h>)
#  define TDM_SIX_ARGS 1
# endif
#endif
#ifndef TDM_SIX_ARGS
# define TDM_SIX_ARGS 0
#endif
#if defined(__has_builtin)
# if __has_builtin(__builtin_amdgcn_tensor_load_to_lds)
#  define HAVE_TDM 1
# endif
# if __has_builtin(__builtin_amdgcn_s_wait_tensorcnt)
#  define HAVE_WAIT_TENSOR 1
# endif
#endif
#ifndef HAVE_TDM
# define HAVE_TDM 0
#endif
#ifndef HAVE_WAIT_TENSOR
# define HAVE_WAIT_TENSOR 0
#endif

__device__ __forceinline__ unsigned short f2bf(float f) {
    unsigned int u = __float_as_uint(f);
    unsigned int r = (u + 0x7FFFu + ((u >> 16) & 1u)) >> 16;   // RNE
    return (unsigned short)r;
}

__device__ __forceinline__ float rmax16(float t) {
    t = fmaxf(t, __shfl_xor(t, 1, 32));
    t = fmaxf(t, __shfl_xor(t, 2, 32));
    t = fmaxf(t, __shfl_xor(t, 4, 32));
    t = fmaxf(t, __shfl_xor(t, 8, 32));
    return t;
}
__device__ __forceinline__ float rsum16(float t) {
    t += __shfl_xor(t, 1, 32);
    t += __shfl_xor(t, 2, 32);
    t += __shfl_xor(t, 4, 32);
    t += __shfl_xor(t, 8, 32);
    return t;
}

__device__ __forceinline__ void wait_tensor0() {
#if HAVE_WAIT_TENSOR
    __builtin_amdgcn_s_wait_tensorcnt(0);
#else
    asm volatile("s_wait_tensorcnt 0x0" ::: "memory");
#endif
}

// ---------------------------------------------------------------------------
// TDM: DMA one 128-row x 64-col bf16 tile (row stride ld elements) from
// global into LDS at lds_off, padding each 128B row by 16B -> stride 72
// elements, matching the compute-side fragment layout.
// D# group0: count=1 | lds_addr | global_addr(57b) | type=2.
// D# group1: data_size=2B, pad_enable, pad_interval=32DW(code 4),
//            pad_amount=4DW(code 3), tensor_dim0/1, tile 64x128, dim0 stride.
// ---------------------------------------------------------------------------
__device__ __forceinline__ void tdm_tile_128x64(unsigned lds_off,
                                                const unsigned short* g, int ld) {
#if HAVE_TDM
    unsigned long long ga = (unsigned long long)g;
    u32x4 g0;
    g0[0] = 1u;                                     // count=1, user descriptor
    g0[1] = lds_off;                                // LDS byte address
    g0[2] = (unsigned)ga;                           // global addr lo
    g0[3] = (unsigned)(ga >> 32) | (2u << 30);      // global addr hi | type=2
    unsigned dim0 = (unsigned)ld;                   // tensor dim0 (elements)
    unsigned dim1 = 1u << 22;                       // generous row bound
    i32x8 g1;
    g1[0] = (int)((1u << 16) | (1u << 20) | (4u << 22) | (3u << 25));
    g1[1] = (int)((dim0 & 0xFFFFu) << 16);          // [63:48] dim0 lo16
    g1[2] = (int)((dim0 >> 16) | ((dim1 & 0xFFFFu) << 16));
    g1[3] = (int)((dim1 >> 16) | (64u << 16));      // tile_dim0 = 64
    g1[4] = (int)128u;                              // tile_dim1 = 128, dim2=0
    g1[5] = (int)(unsigned)ld;                      // tensor_dim0_stride lo32
    g1[6] = 0;
    g1[7] = 0;
    i32x4 gz = {0, 0, 0, 0};
#if TDM_SIX_ARGS
    i32x8 z8 = {0, 0, 0, 0, 0, 0, 0, 0};
    __builtin_amdgcn_tensor_load_to_lds(g0, g1, gz, gz, z8, 0);
#else
    __builtin_amdgcn_tensor_load_to_lds(g0, g1, gz, gz, 0);
#endif
#else
    (void)lds_off; (void)g; (void)ld;
#endif
}

// ---------------------------------------------------------------------------
// f32 [K][N] -> bf16 [N][K] transpose-convert (weights, once per launch)
// ---------------------------------------------------------------------------
__global__ __launch_bounds__(256) void convert_transpose_kernel(
    const float* __restrict__ in, unsigned short* __restrict__ out,
    int K, int N) {
    __shared__ unsigned short tile[32][33];
    const int nt0 = blockIdx.x * 32, kt0 = blockIdx.y * 32;
    const int tx = threadIdx.x & 31, ty = threadIdx.x >> 5;   // 32 x 8
    #pragma unroll
    for (int i = 0; i < 4; ++i) {
        int k = ty + i * 8;
        tile[k][tx] = f2bf(in[(size_t)(kt0 + k) * N + nt0 + tx]);
    }
    __syncthreads();
    #pragma unroll
    for (int i = 0; i < 4; ++i) {
        int n = ty + i * 8;
        out[(size_t)(nt0 + n) * K + kt0 + tx] = tile[tx][n];
    }
}

// ---------------------------------------------------------------------------
// h = tok_emb[x] + pos_emb[t]
// ---------------------------------------------------------------------------
__global__ __launch_bounds__(256) void embed_kernel(
    const int* __restrict__ x, const float* __restrict__ tok,
    const float* __restrict__ pos, float* __restrict__ h) {
    int base = (blockIdx.x * 256 + threadIdx.x) * 4;
    int row  = base >> 10;
    int col  = base & 1023;
    int tknn = x[row];
    int t    = row & (Tdim - 1);
    f32x4 a = *(const f32x4*)(tok + (size_t)tknn * Edim + col);
    f32x4 p = *(const f32x4*)(pos + (size_t)t * Edim + col);
    f32x4 o = a + p;
    *(f32x4*)(h + (size_t)row * Edim + col) = o;
}

// ---------------------------------------------------------------------------
// LayerNorm: f32 in [rows,1024] -> bf16 out (feeds WMMA GEMMs)
// ---------------------------------------------------------------------------
__global__ __launch_bounds__(256) void ln_kernel(
    const float* __restrict__ X, const float* __restrict__ g,
    const float* __restrict__ be, unsigned short* __restrict__ Y) {
    __shared__ float red[256];
    const int r = blockIdx.x, t = threadIdx.x;
    const float* x = X + (size_t)r * Edim;
    float v[4];
    #pragma unroll
    for (int i = 0; i < 4; ++i) v[i] = x[t + 256 * i];
    float s = v[0] + v[1] + v[2] + v[3];
    red[t] = s; __syncthreads();
    for (int o = 128; o > 0; o >>= 1) { if (t < o) red[t] += red[t + o]; __syncthreads(); }
    float mean = red[0] * (1.0f / Edim);
    __syncthreads();
    float q = 0.f;
    #pragma unroll
    for (int i = 0; i < 4; ++i) { float d = v[i] - mean; q += d * d; }
    red[t] = q; __syncthreads();
    for (int o = 128; o > 0; o >>= 1) { if (t < o) red[t] += red[t + o]; __syncthreads(); }
    float rst = rsqrtf(red[0] * (1.0f / Edim) + 1e-5f);
    #pragma unroll
    for (int i = 0; i < 4; ++i) {
        int c = t + 256 * i;
        Y[(size_t)r * Edim + c] = f2bf((v[i] - mean) * rst * g[c] + be[c]);
    }
}

// ---------------------------------------------------------------------------
// bf16 WMMA GEMM: C[M,N] = act(A[M,K] @ W^T[N,K]^T + bias) (+resid).
// WT is the pre-transposed weight [N][K]. 256 thr, tile 128x128xBK64,
// 8 waves as 2(m)x4(n), each wave 64x32 = 8 accumulators. Both A and WT
// tiles are row-major 128x64 -> staged by the TDM (padded to stride 72);
// fragments load as 2x ds_load_b128 each.
// ACT: 0=none 1=exact GELU.  RES: add f32 resid.  OBF: bf16 output.
// ---------------------------------------------------------------------------
__device__ __forceinline__ void stage_tile_manual(
    unsigned short* dst, const unsigned short* src, int ld, int tid) {
    #pragma unroll
    for (int r = 0; r < 4; ++r) {
        int idx = tid + r * 256;
        int ar = idx >> 3, ac = (idx & 7) << 3;
        u32x4 d = *(const u32x4*)(src + (size_t)ar * ld + ac);
        *(u32x4*)(dst + ar * 72 + ac) = d;
    }
}

template <int ACT, int RES, int OBF>
__global__ __launch_bounds__(256) void gemm_kernel(
    const unsigned short* __restrict__ A, const unsigned short* __restrict__ WT,
    const float* __restrict__ bias, const float* __restrict__ resid,
    void* __restrict__ Cv, int M, int N, int K) {
    __shared__ unsigned short As[128 * 72];    // [row][k], stride 72
    __shared__ unsigned short Bs[128 * 72];    // [n][k],  stride 72
    const int tid = threadIdx.x;
    const int bm = blockIdx.y, bn = blockIdx.x;
    const int lane = tid & 31, wid = tid >> 5;
    const int wm = wid & 1, wn = wid >> 1;
    const int rowA  = lane & 15;
    const int kbase = (lane & 16) ? 8 : 0;     // 16-bit WMMA operand K striping
    v8f acc[4][2] = {};
    const int nkt = K >> 6;
#if HAVE_TDM
    const unsigned AsOff = (unsigned)(size_t)&As[0];
    const unsigned BsOff = (unsigned)(size_t)&Bs[0];
#endif
    for (int kt = 0; kt < nkt; ++kt) {
        __syncthreads();
#if HAVE_TDM
        if (wid == 0) {                        // wave-uniform: TDM ignores EXEC
            tdm_tile_128x64(AsOff, A  + (size_t)(bm * 128) * K + kt * 64, K);
            tdm_tile_128x64(BsOff, WT + (size_t)(bn * 128) * K + kt * 64, K);
            wait_tensor0();
        }
#else
        stage_tile_manual(As, A  + (size_t)(bm * 128) * K + kt * 64, K, tid);
        stage_tile_manual(Bs, WT + (size_t)(bn * 128) * K + kt * 64, K, tid);
#endif
        __syncthreads();
        #pragma unroll
        for (int ks = 0; ks < 64; ks += 32) {
            Frag16 af[4], bf[2];
            #pragma unroll
            for (int i = 0; i < 4; ++i) {
                const unsigned short* p = As + (wm * 64 + i * 16 + rowA) * 72 + ks + kbase;
                af[i].q2[0] = *(const u32x4*)p;
                af[i].q2[1] = *(const u32x4*)(p + 16);
            }
            #pragma unroll
            for (int j = 0; j < 2; ++j) {
                const unsigned short* p = Bs + (wn * 32 + j * 16 + rowA) * 72 + ks + kbase;
                bf[j].q2[0] = *(const u32x4*)p;
                bf[j].q2[1] = *(const u32x4*)(p + 16);
            }
            #pragma unroll
            for (int i = 0; i < 4; ++i)
                #pragma unroll
                for (int j = 0; j < 2; ++j)
                    acc[i][j] = __builtin_amdgcn_wmma_f32_16x16x32_bf16(
                        false, af[i].v, false, bf[j].v, (short)0, acc[i][j], false, false);
        }
    }
    const int rhalf = (lane & 16) ? 8 : 0;
    #pragma unroll
    for (int j = 0; j < 2; ++j) {
        int col = bn * 128 + wn * 32 + j * 16 + (lane & 15);
        float bv = bias[col];
        #pragma unroll
        for (int i = 0; i < 4; ++i) {
            int r0 = bm * 128 + wm * 64 + i * 16 + rhalf;
            #pragma unroll
            for (int e = 0; e < 8; ++e) {
                float v = acc[i][j][e] + bv;
                if (ACT == 1) v = 0.5f * v * (1.0f + erff(v * 0.70710678118f));
                size_t off = (size_t)(r0 + e) * N + col;
                if (RES) v += resid[off];
                if (OBF) ((unsigned short*)Cv)[off] = f2bf(v);
                else     ((float*)Cv)[off] = v;
            }
        }
    }
}

// ---------------------------------------------------------------------------
// Flash attention, causal. One wave per 16-row Q tile per (b,h).
// QK^T: 2 chained WMMAs over D=64.  PV: 4 WMMAs over 32-key blocks.
// Online softmax with half-wave shuffle reductions (matches C-frag striping).
// ---------------------------------------------------------------------------
__global__ __launch_bounds__(32) void attn_kernel(
    const unsigned short* __restrict__ Q, const unsigned short* __restrict__ Kb,
    const unsigned short* __restrict__ Vb, unsigned short* __restrict__ O) {
    __shared__ unsigned short Pst[16 * 32];     // P tile bf16
    __shared__ unsigned short VsT[64 * 40];     // V^T: [d][k]
    const int lane = threadIdx.x;
    const int q0 = blockIdx.x * 16, hh = blockIdx.y, bb = blockIdx.z;
    const int half = (lane >> 4) & 1, nn = lane & 15;
    const int kb0 = half * 8;
    const float SCALE = 0.125f;                  // 1/sqrt(64)
    const size_t basebh = ((size_t)bb * Tdim) * Edim + hh * Ddim;

    Frag16 qa[2];
    #pragma unroll
    for (int df = 0; df < 2; ++df) {             // Q tile 16 x 64, A-fragment layout
        const unsigned short* p = Q + basebh + (size_t)(q0 + nn) * Edim + df * 32 + kb0;
        qa[df].q2[0] = *(const u32x4*)p;
        qa[df].q2[1] = *(const u32x4*)(p + 16);
    }

    v8f acc[4] = {};
    float mrow[8], lrow[8];
    #pragma unroll
    for (int i = 0; i < 8; ++i) { mrow[i] = -3.0e38f; lrow[i] = 0.f; }

    for (int k0 = 0; k0 <= q0 + 15; k0 += 32) {
        __syncthreads();
        {   // stage V^T: lane loads one key row, scatters to [d][k]
            const unsigned short* vp = Vb + basebh + (size_t)(k0 + lane) * Edim;
            #pragma unroll
            for (int c = 0; c < 8; ++c) {
                U4U8 t; t.q = *(const u32x4*)(vp + c * 8);
                #pragma unroll
                for (int i = 0; i < 8; ++i) VsT[(c * 8 + i) * 40 + lane] = t.s[i];
            }
        }
        v8f s[2];
        #pragma unroll
        for (int kc = 0; kc < 2; ++kc) {         // S = Q K^T, two 16x16 tiles
            v8f sc = {};
            #pragma unroll
            for (int df = 0; df < 2; ++df) {
                Frag16 kf;
                const unsigned short* kp =
                    Kb + basebh + (size_t)(k0 + 16 * kc + nn) * Edim + df * 32 + kb0;
                kf.q2[0] = *(const u32x4*)kp;
                kf.q2[1] = *(const u32x4*)(kp + 16);
                sc = __builtin_amdgcn_wmma_f32_16x16x32_bf16(
                    false, qa[df].v, false, kf.v, (short)0, sc, false, false);
            }
            s[kc] = sc;
        }
        float alpha[8];
        #pragma unroll
        for (int i = 0; i < 8; ++i) {            // online softmax update per row
            int qrow = q0 + i + 8 * half;
            float s0 = s[0][i] * SCALE, s1 = s[1][i] * SCALE;
            if (k0 + nn      > qrow) s0 = -3.0e38f;
            if (k0 + 16 + nn > qrow) s1 = -3.0e38f;
            float mn = fmaxf(mrow[i], rmax16(fmaxf(s0, s1)));
            alpha[i] = __expf(mrow[i] - mn);
            float p0 = __expf(s0 - mn), p1 = __expf(s1 - mn);
            Pst[(i + 8 * half) * 32 + nn]      = f2bf(p0);
            Pst[(i + 8 * half) * 32 + 16 + nn] = f2bf(p1);
            lrow[i] = lrow[i] * alpha[i] + rsum16(p0 + p1);
            mrow[i] = mn;
        }
        #pragma unroll
        for (int nf = 0; nf < 4; ++nf)
            #pragma unroll
            for (int i = 0; i < 8; ++i) acc[nf][i] *= alpha[i];
        __syncthreads();
        Frag16 pf;
        const unsigned short* pp = Pst + nn * 32 + kb0;
        pf.q2[0] = *(const u32x4*)pp;
        pf.q2[1] = *(const u32x4*)(pp + 16);
        #pragma unroll
        for (int nf = 0; nf < 4; ++nf) {         // O += P V
            Frag16 vf;
            const unsigned short* vq = VsT + (nf * 16 + nn) * 40 + kb0;
            vf.q2[0] = *(const u32x4*)vq;
            vf.q2[1] = *(const u32x4*)(vq + 16);
            acc[nf] = __builtin_amdgcn_wmma_f32_16x16x32_bf16(
                false, pf.v, false, vf.v, (short)0, acc[nf], false, false);
        }
    }
    float rl[8];
    #pragma unroll
    for (int i = 0; i < 8; ++i) rl[i] = 1.0f / lrow[i];
    #pragma unroll
    for (int nf = 0; nf < 4; ++nf)
        #pragma unroll
        for (int i = 0; i < 8; ++i)
            O[basebh + (size_t)(q0 + i + 8 * half) * Edim + nf * 16 + nn] =
                f2bf(acc[nf][i] * rl[i]);
}

// ---------------------------------------------------------------------------
extern "C" void kernel_launch(void* const* d_in, const int* in_sizes, int n_in,
                              void* d_out, int out_size, void* d_ws, size_t ws_size,
                              hipStream_t stream) {
    (void)in_sizes; (void)n_in; (void)out_size; (void)ws_size;
    const int*   x    = (const int*)d_in[0];
    const float* tok  = (const float*)d_in[1];
    const float* pos  = (const float*)d_in[2];
    const float* Wq   = (const float*)d_in[3];  const float* bq = (const float*)d_in[4];
    const float* Wk   = (const float*)d_in[5];  const float* bk = (const float*)d_in[6];
    const float* Wv   = (const float*)d_in[7];  const float* bv = (const float*)d_in[8];
    const float* Wo   = (const float*)d_in[9];  const float* bo = (const float*)d_in[10];
    const float* ln1g = (const float*)d_in[11]; const float* ln1b = (const float*)d_in[12];
    const float* W1   = (const float*)d_in[13]; const float* b1 = (const float*)d_in[14];
    const float* W2   = (const float*)d_in[15]; const float* b2 = (const float*)d_in[16];
    const float* ln2g = (const float*)d_in[17]; const float* ln2b = (const float*)d_in[18];
    const float* lnfg = (const float*)d_in[19]; const float* lnfb = (const float*)d_in[20];
    const float* Wout = (const float*)d_in[21]; const float* bout = (const float*)d_in[22];

    char* ws = (char*)d_ws;
    size_t off = 0;
    auto take = [&](size_t bytes) -> char* {
        char* p = ws + off;
        off += (bytes + 255) & ~(size_t)255;
        return p;
    };
    float*          h    = (float*)take((size_t)Mrows * Edim * 4);
    unsigned short* xn   = (unsigned short*)take((size_t)Mrows * Edim * 2);
    unsigned short* qb   = (unsigned short*)take((size_t)Mrows * Edim * 2);
    unsigned short* kbuf = (unsigned short*)take((size_t)Mrows * Edim * 2);
    unsigned short* vbuf = (unsigned short*)take((size_t)Mrows * Edim * 2);
    unsigned short* ob   = (unsigned short*)take((size_t)Mrows * Edim * 2);
    unsigned short* ub   = (unsigned short*)take((size_t)Mrows * Fdim * 2);
    unsigned short* wqb  = (unsigned short*)take((size_t)Edim * Edim * 2);
    unsigned short* wkb  = (unsigned short*)take((size_t)Edim * Edim * 2);
    unsigned short* wvb  = (unsigned short*)take((size_t)Edim * Edim * 2);
    unsigned short* wob  = (unsigned short*)take((size_t)Edim * Edim * 2);
    unsigned short* w1b  = (unsigned short*)take((size_t)Edim * Fdim * 2);
    unsigned short* w2b  = (unsigned short*)take((size_t)Fdim * Edim * 2);
    unsigned short* woutb = (unsigned short*)take((size_t)Edim * Vdim * 2);

    // transpose-convert grids: dim3(N/32, K/32) for original W[K][N]
    dim3 gtEE(Edim / 32, Edim / 32);
    dim3 gtEF(Fdim / 32, Edim / 32);
    dim3 gtFE(Edim / 32, Fdim / 32);
    dim3 gtEV(Vdim / 32, Edim / 32);
    convert_transpose_kernel<<<gtEV, 256, 0, stream>>>(Wout, woutb, Edim, Vdim);
    embed_kernel<<<(Mrows * Edim) / 1024, 256, 0, stream>>>(x, tok, pos, h);

    const int nEE = Edim * Edim, nEF = Edim * Fdim;
    dim3 gE(Edim / 128, Mrows / 128);     // (8, 32)
    dim3 gF(Fdim / 128, Mrows / 128);     // (32, 32)
    for (int l = 0; l < Ldim; ++l) {
        convert_transpose_kernel<<<gtEE, 256, 0, stream>>>(Wq + (size_t)l * nEE, wqb, Edim, Edim);
        convert_transpose_kernel<<<gtEE, 256, 0, stream>>>(Wk + (size_t)l * nEE, wkb, Edim, Edim);
        convert_transpose_kernel<<<gtEE, 256, 0, stream>>>(Wv + (size_t)l * nEE, wvb, Edim, Edim);
        convert_transpose_kernel<<<gtEE, 256, 0, stream>>>(Wo + (size_t)l * nEE, wob, Edim, Edim);
        convert_transpose_kernel<<<gtEF, 256, 0, stream>>>(W1 + (size_t)l * nEF, w1b, Edim, Fdim);
        convert_transpose_kernel<<<gtFE, 256, 0, stream>>>(W2 + (size_t)l * nEF, w2b, Fdim, Edim);

        ln_kernel<<<Mrows, 256, 0, stream>>>(h, ln1g + l * Edim, ln1b + l * Edim, xn);
        gemm_kernel<0, 0, 1><<<gE, 256, 0, stream>>>(xn, wqb, bq + l * Edim, nullptr, qb,   Mrows, Edim, Edim);
        gemm_kernel<0, 0, 1><<<gE, 256, 0, stream>>>(xn, wkb, bk + l * Edim, nullptr, kbuf, Mrows, Edim, Edim);
        gemm_kernel<0, 0, 1><<<gE, 256, 0, stream>>>(xn, wvb, bv + l * Edim, nullptr, vbuf, Mrows, Edim, Edim);
        attn_kernel<<<dim3(Tdim / 16, Hn, Bsz), 32, 0, stream>>>(qb, kbuf, vbuf, ob);
        gemm_kernel<0, 1, 0><<<gE, 256, 0, stream>>>(ob, wob, bo + l * Edim, h, h, Mrows, Edim, Edim);
        ln_kernel<<<Mrows, 256, 0, stream>>>(h, ln2g + l * Edim, ln2b + l * Edim, xn);
        gemm_kernel<1, 0, 1><<<gF, 256, 0, stream>>>(xn, w1b, b1 + l * Fdim, nullptr, ub, Mrows, Fdim, Edim);
        gemm_kernel<0, 1, 0><<<gE, 256, 0, stream>>>(ub, w2b, b2 + l * Edim, h, h, Mrows, Edim, Fdim);
    }
    ln_kernel<<<Mrows, 256, 0, stream>>>(h, lnfg, lnfb, xn);
    gemm_kernel<0, 0, 0><<<dim3(Vdim / 128, Mrows / 128), 256, 0, stream>>>(
        xn, woutb, bout, nullptr, (float*)d_out, Mrows, Vdim, Edim);
}
